// GNN_65060164600376
// MI455X (gfx1250) — matrix operand order repaired and checked by reading, not compiled, not run on hardware
//
#include <hip/hip_runtime.h>
#include <hip/hip_bf16.h>

// ---------------------------------------------------------------------------
// GNN (TransformerConv-style) forward for MI455X (gfx1250, wave32, WMMA).
//
//  * e_feat @ We  ==  (emb @ We)[edge_attr]  -> 30x128 table GEMM + gather
//    (kills the 800k-row edge GEMM).
//  * Node GEMMs [50000,128]x[128,128] on v_wmma_f32_16x16x32_f16 (f32 accum),
//    BN+ReLU fused into the A-tile load; persistent blocks stage the weight
//    matrix once (via the Tensor Data Mover when the builtin is available)
//    and grid-stride over row tiles with prefetch of the next tile.
//  * Segment softmax via order-preserving u32-encoded atomicMax + f32
//    atomicAdd (native L2 atomics); node tensors (25.6MB) are L2-resident.
// ---------------------------------------------------------------------------

#define N_NODES 50000
#define N_EDGES 800000
#define HID 128
#define G_GRAPHS 500
#define LAYERS 4
#define EPSBN 1e-5f

typedef _Float16 v8h  __attribute__((ext_vector_type(8)));
typedef _Float16 v16h __attribute__((ext_vector_type(16)));
typedef float    v8f  __attribute__((ext_vector_type(8)));
typedef unsigned u32x4 __attribute__((ext_vector_type(4)));
typedef int      i32x4 __attribute__((ext_vector_type(4)));
typedef int      i32x8 __attribute__((ext_vector_type(8)));

#if __has_builtin(__builtin_amdgcn_tensor_load_to_lds) && __has_builtin(__builtin_amdgcn_s_wait_tensorcnt)
#define GEMM_TDM 1
#else
#define GEMM_TDM 0
#endif

__device__ __forceinline__ v16h cat8(v8h lo, v8h hi) {
  return __builtin_shufflevector(lo, hi, 0,1,2,3,4,5,6,7,8,9,10,11,12,13,14,15);
}

__device__ __forceinline__ unsigned encMax(float f) {
  unsigned u = __float_as_uint(f);
  return (u & 0x80000000u) ? ~u : (u | 0x80000000u);
}
__device__ __forceinline__ float decMax(unsigned k) {
  unsigned u = (k & 0x80000000u) ? (k & 0x7fffffffu) : ~k;
  return __uint_as_float(u);
}

#define TP 136  // LDS tile pitch in halves: 272B row -> fragment reads spread banks

// C[M,128] = act(bn?(A)) @ W[128,128] + bias ; optional second destination.
// mode: 0 = plain A, 1 = BN on A-load, 2 = BN+ReLU on A-load.
// Persistent blocks: W is staged/transposed once, then grid-stride over tiles.
__global__ __launch_bounds__(128)
void gemm128_wmma(const float* __restrict__ A, const float* __restrict__ Wm,
                  const float* __restrict__ bias,
                  const float* __restrict__ meanv, const float* __restrict__ rstdv,
                  const float* __restrict__ gam,   const float* __restrict__ bet,
                  int mode, float* __restrict__ C, float* __restrict__ C2,
                  int M, int numTiles)
{
#if GEMM_TDM
  // [0, 64KB)      : raw f32 W tile landed by the TDM (lds_addr = 0)
  //                  -> reused as the f16 A tile once W is transposed
  // [64KB, +34816) : f16 transposed W
  __shared__ __align__(16) char smem[65536 + 128 * TP * 2];
  float*    Wst = (float*)smem;
  _Float16* Wt  = (_Float16*)(smem + 65536);
  _Float16* At  = (_Float16*)smem;
#else
  __shared__ __align__(16) char smem[(64 + 128) * TP * 2];
  _Float16* At = (_Float16*)smem;
  _Float16* Wt = (_Float16*)(smem + 64 * TP * 2);
#endif

#if GEMM_TDM
  if ((threadIdx.x >> 5) == 0) {
    // Tensor Data Mover: DMA the 128x128 f32 weight matrix (one 2D tile,
    // contiguous 64KB) into LDS offset 0. Descriptor per CDNA5 ISA 8.3/8.4.
    unsigned long long ga = (unsigned long long)(const void*)Wm;
    u32x4 g0;
    g0[0] = 1u;                                   // count=1 (valid user D#)
    g0[1] = 0u;                                   // lds_addr = 0
    g0[2] = (unsigned)ga;                         // global_addr[31:0]
    g0[3] = (unsigned)((ga >> 32) & 0x01FFFFFFull) | (2u << 30); // addr[56:32]|type=2
    i32x8 g1;
    g1[0] = 0x00020000;                           // data_size=2 (4 bytes)
    g1[1] = (int)(128u << 16);                    // tensor_dim0 = 128
    g1[2] = (int)(128u << 16);                    // tensor_dim1 = 128
    g1[3] = (int)(128u << 16);                    // tile_dim0 = 128
    g1[4] = 128;                                  // tile_dim1 = 128
    g1[5] = 128;                                  // tensor_dim0_stride = 128
    g1[6] = 0;
    g1[7] = 0;
    i32x4 z4 = {0, 0, 0, 0};
#if defined(__clang_major__) && (__clang_major__ >= 23)
    i32x8 z8 = {0, 0, 0, 0, 0, 0, 0, 0};
    __builtin_amdgcn_tensor_load_to_lds(g0, g1, z4, z4, z8, 0);
#else
    __builtin_amdgcn_tensor_load_to_lds(g0, g1, z4, z4, 0);
#endif
    __builtin_amdgcn_s_wait_tensorcnt(0);
  }
  __syncthreads();
  for (int i = threadIdx.x; i < 128 * 128; i += 128) {
    int k = i >> 7, n = i & 127;
    Wt[n * TP + k] = (_Float16)Wst[i];
  }
#else
  for (int i = threadIdx.x; i < 128 * 128; i += 128) {
    int k = i >> 7, n = i & 127;
    Wt[n * TP + k] = (_Float16)Wm[i];
  }
#endif

  const int lane = threadIdx.x & 31;
  const int wave = threadIdx.x >> 5;
  const int rm   = lane & 15;   // row (A) / column (B) within 16x16 tile
  const int hs   = lane >> 4;   // half-wave selector

  for (int tile = blockIdx.x; tile < numTiles; tile += gridDim.x) {
    const int rowBase = tile * 64;
    __syncthreads();  // At reuse across iterations; also orders Wt writes->reads

    // Stage 64-row A tile with fused BN(+ReLU), f32 -> f16.
    for (int i = threadIdx.x; i < 64 * 32; i += 128) {
      int r = i >> 5, c0 = (i & 31) * 4;
      int gr = rowBase + r;
      float v[4] = {0.f, 0.f, 0.f, 0.f};
      if (gr < M) {
        const float4 t = *(const float4*)(A + (size_t)gr * 128 + c0);
        v[0] = t.x; v[1] = t.y; v[2] = t.z; v[3] = t.w;
        if (mode) {
          #pragma unroll
          for (int j = 0; j < 4; ++j) {
            int c = c0 + j;
            v[j] = gam[c] * (v[j] - meanv[c]) * rstdv[c] + bet[c];
            if (mode == 2) v[j] = fmaxf(v[j], 0.f);
          }
        }
      }
      #pragma unroll
      for (int j = 0; j < 4; ++j) At[r * TP + c0 + j] = (_Float16)v[j];
    }
    // Prefetch next tile of A into L2 while this tile computes.
    {
      int tnext = tile + gridDim.x;
      if (tnext < numTiles) {
        const float* p = A + (size_t)tnext * 64 * 128 + (size_t)threadIdx.x * 64;
        __builtin_prefetch(p, 0, 1);
        __builtin_prefetch(p + 32, 0, 1);
      }
    }
    __syncthreads();

    // A fragments for this wave's 16 rows (K = 0..127, 4 steps of 32).
    v16h aF[4];
    #pragma unroll
    for (int kt = 0; kt < 4; ++kt) {
      const _Float16* p = &At[(wave * 16 + rm) * TP + kt * 32 + hs * 8];
      v8h lo = *(const v8h*)p;
      v8h hi = *(const v8h*)(p + 16);
      aF[kt] = cat8(lo, hi);
    }

    const bool fullTile = (rowBase + 64 <= M);
    #pragma unroll
    for (int nt = 0; nt < 8; ++nt) {
      v8f acc = {0.f, 0.f, 0.f, 0.f, 0.f, 0.f, 0.f, 0.f};
      #pragma unroll
      for (int kt = 0; kt < 4; ++kt) {
        const _Float16* p = &Wt[(nt * 16 + rm) * TP + kt * 32 + hs * 8];
        v8h lo = *(const v8h*)p;
        v8h hi = *(const v8h*)(p + 16);
        v16h bF = cat8(lo, hi);
        acc = __builtin_amdgcn_wmma_f32_16x16x32_f16(
            false, aF[kt], false, bF, (short)0, acc, false, false);
      }
      const int col = nt * 16 + rm;
      const float bv = bias[col];
      const size_t base = (size_t)(rowBase + wave * 16 + hs * 8) * 128 + col;
      if (fullTile) {                       // uniform: unconditional stores
        if (C2) {                           // uniform
          #pragma unroll
          for (int r8 = 0; r8 < 8; ++r8) {
            float o = acc[r8] + bv;
            C[base + (size_t)r8 * 128]  = o;
            C2[base + (size_t)r8 * 128] = o;
          }
        } else {
          #pragma unroll
          for (int r8 = 0; r8 < 8; ++r8)
            C[base + (size_t)r8 * 128] = acc[r8] + bv;
        }
      } else {
        #pragma unroll
        for (int r8 = 0; r8 < 8; ++r8) {
          int m = rowBase + wave * 16 + hs * 8 + r8;
          if (m < M) {
            float o = acc[r8] + bv;
            C[(size_t)m * 128 + col] = o;
            if (C2) C2[(size_t)m * 128 + col] = o;
          }
        }
      }
    }
  }
}

// T[30,128] = A[30,128] @ W[128,128] + bias (tiny; one block).
__global__ __launch_bounds__(128)
void table_gemm(const float* __restrict__ A, const float* __restrict__ Wm,
                const float* __restrict__ bias, float* __restrict__ T, int Mrows)
{
  __shared__ float As[30 * 128];
  for (int i = threadIdx.x; i < Mrows * 128; i += 128) As[i] = A[i];
  __syncthreads();
  int c = threadIdx.x;
  for (int r = 0; r < Mrows; ++r) {
    float acc = bias[c];
    #pragma unroll 8
    for (int k = 0; k < 128; ++k) acc = fmaf(As[r * 128 + k], Wm[k * 128 + c], acc);
    T[r * 128 + c] = acc;
  }
}

__global__ void lookup_rows(const int* __restrict__ idx, const float* __restrict__ T,
                            float* __restrict__ out, int n)
{
  int id = blockIdx.x * blockDim.x + threadIdx.x;
  if (id < n * 128) out[id] = T[idx[id >> 7] * 128 + (id & 127)];
}

__global__ void fill_k(unsigned* __restrict__ p, unsigned v, int n)
{
  int id = blockIdx.x * blockDim.x + threadIdx.x;
  if (id < n) p[id] = v;
}

__global__ __launch_bounds__(128)
void colstats_k(const float* __restrict__ X, int M,
                float* __restrict__ sum, float* __restrict__ sumsq)
{
  int c = threadIdx.x;
  float s = 0.f, s2 = 0.f;
  for (int r = blockIdx.x; r < M; r += gridDim.x) {
    float v = X[(size_t)r * 128 + c];
    s += v; s2 += v * v;
  }
  atomicAdd(&sum[c], s);
  atomicAdd(&sumsq[c], s2);
}

__global__ void finalize_k(const float* __restrict__ sum, const float* __restrict__ sumsq,
                           float invM, float* __restrict__ meanv, float* __restrict__ rstdv)
{
  int c = threadIdx.x;
  float m = sum[c] * invM;
  float v = sumsq[c] * invM - m * m;
  meanv[c] = m;
  rstdv[c] = rsqrtf(v + EPSBN);
}

// One wave per edge: alpha[e,h] = dot(Q[dst], K[src]+Te[attr]) / sqrt(32),
// running atomic segment-max per (dst, head).
__global__ void edge_alpha_k(const float* __restrict__ Q, const float* __restrict__ K,
                             const float* __restrict__ Te,
                             const int* __restrict__ src, const int* __restrict__ dst,
                             const int* __restrict__ attr,
                             float* __restrict__ alpha, unsigned* __restrict__ mx, int E)
{
  int t = blockIdx.x * blockDim.x + threadIdx.x;
  int j = t >> 5, lane = t & 31;
  if (j >= E) return;
  int s = src[j], d = dst[j], a = attr[j];
  const float4 q  = *(const float4*)(Q  + (size_t)d * 128 + lane * 4);
  const float4 kk = *(const float4*)(K  + (size_t)s * 128 + lane * 4);
  const float4 te = *(const float4*)(Te + (size_t)a * 128 + lane * 4);
  float p = q.x*(kk.x+te.x) + q.y*(kk.y+te.y) + q.z*(kk.z+te.z) + q.w*(kk.w+te.w);
  p += __shfl_xor(p, 1, 32);
  p += __shfl_xor(p, 2, 32);
  p += __shfl_xor(p, 4, 32);   // lanes 8h..8h+7 now hold head-h sum
  if ((lane & 7) == 0) {
    int hh = lane >> 3;
    float av = p * 0.17677669529663689f;   // 1/sqrt(32)
    alpha[(size_t)j * 4 + hh] = av;
    atomicMax(&mx[(size_t)d * 4 + hh], encMax(av));
  }
}

__global__ void edge_exp_k(const int* __restrict__ dst, const unsigned* __restrict__ mx,
                           float* __restrict__ alpha, float* __restrict__ den, int E)
{
  int t = blockIdx.x * blockDim.x + threadIdx.x;
  if (t >= E * 4) return;
  int j = t >> 2, hh = t & 3;
  int d = dst[j];
  float ex = __expf(alpha[t] - decMax(mx[(size_t)d * 4 + hh]));
  alpha[t] = ex;
  atomicAdd(&den[(size_t)d * 4 + hh], ex);
}

// One wave per edge: agg[dst] += softmax * (V[src] + Te[attr]).
__global__ void edge_agg_k(const float* __restrict__ V, const float* __restrict__ Te,
                           const int* __restrict__ src, const int* __restrict__ dst,
                           const int* __restrict__ attr,
                           const float* __restrict__ ex, const float* __restrict__ den,
                           float* __restrict__ agg, int E)
{
  int t = blockIdx.x * blockDim.x + threadIdx.x;
  int j = t >> 5, lane = t & 31;
  if (j >= E) return;
  int s = src[j], d = dst[j], a = attr[j];
  int hh = lane >> 3;
  float w = ex[(size_t)j * 4 + hh] / (den[(size_t)d * 4 + hh] + 1e-16f);
  const float4 v  = *(const float4*)(V  + (size_t)s * 128 + lane * 4);
  const float4 te = *(const float4*)(Te + (size_t)a * 128 + lane * 4);
  float* pp = agg + (size_t)d * 128 + lane * 4;
  atomicAdd(pp + 0, w * (v.x + te.x));
  atomicAdd(pp + 1, w * (v.y + te.y));
  atomicAdd(pp + 2, w * (v.z + te.z));
  atomicAdd(pp + 3, w * (v.w + te.w));
}

__global__ void bn_relu_res_k(const float* __restrict__ X,
                              const float* __restrict__ meanv, const float* __restrict__ rstdv,
                              const float* __restrict__ g, const float* __restrict__ b,
                              const float* __restrict__ x0, float* __restrict__ h, int n)
{
  int id = blockIdx.x * blockDim.x + threadIdx.x;
  if (id >= n * 128) return;
  int c = id & 127;
  float v = g[c] * (X[id] - meanv[c]) * rstdv[c] + b[c];
  v = fmaxf(v, 0.f);
  h[id] = 0.9f * v + 0.1f * x0[id];
}

__global__ void pool_sum_k(const float* __restrict__ h, const int* __restrict__ batch,
                           float* __restrict__ pooled, float* __restrict__ cnt, int n)
{
  int t = blockIdx.x * blockDim.x + threadIdx.x;
  int j = t >> 5, lane = t & 31;
  if (j >= n) return;
  int bg = batch[j];
  const float4 hv = *(const float4*)(h + (size_t)j * 128 + lane * 4);
  float* pp = pooled + (size_t)bg * 128 + lane * 4;
  atomicAdd(pp + 0, hv.x);
  atomicAdd(pp + 1, hv.y);
  atomicAdd(pp + 2, hv.z);
  atomicAdd(pp + 3, hv.w);
  if (lane == 0) atomicAdd(&cnt[bg], 1.0f);
}

__global__ void pool_div_k(float* __restrict__ pooled, const float* __restrict__ cnt, int g)
{
  int id = blockIdx.x * blockDim.x + threadIdx.x;
  if (id < g * 128) pooled[id] /= fmaxf(cnt[id >> 7], 1.0f);
}

__global__ __launch_bounds__(128)
void out_final_k(const float* __restrict__ T2,
                 const float* __restrict__ meanv, const float* __restrict__ rstdv,
                 const float* __restrict__ g, const float* __restrict__ b,
                 const float* __restrict__ W2, const float* __restrict__ b2,
                 float* __restrict__ out)
{
  __shared__ float red[128];
  int gi = blockIdx.x, c = threadIdx.x;
  float v = T2[(size_t)gi * 128 + c];
  v = g[c] * (v - meanv[c]) * rstdv[c] + b[c];
  v = fmaxf(v, 0.f) * W2[c];
  red[c] = v;
  __syncthreads();
  for (int o = 64; o > 0; o >>= 1) {
    if (c < o) red[c] += red[c + o];
    __syncthreads();
  }
  if (c == 0) out[gi] = red[0] + b2[0];
}

extern "C" void kernel_launch(void* const* d_in, const int* in_sizes, int n_in,
                              void* d_out, int out_size, void* d_ws, size_t ws_size,
                              hipStream_t stream)
{
  (void)in_sizes; (void)n_in; (void)out_size; (void)ws_size;
  const int N = N_NODES, E = N_EDGES, G = G_GRAPHS;

  // ---- inputs (jax tree_flatten order: dict keys sorted) ----
  const int* xs    = (const int*)d_in[0];
  const int* src   = (const int*)d_in[1];
  const int* dst   = src + E;
  const int* eattr = (const int*)d_in[2];
  const int* batch = (const int*)d_in[3];
  const float* bnend_b = (const float*)d_in[4];
  const float* bnend_g = (const float*)d_in[5];
  const float* bns_b[LAYERS], *bns_g[LAYERS];
  for (int l = 0; l < LAYERS; ++l) { bns_b[l] = (const float*)d_in[6 + 2*l];
                                     bns_g[l] = (const float*)d_in[7 + 2*l]; }
  const float *We[LAYERS], *Wk[LAYERS], *Wq[LAYERS], *Wv[LAYERS];
  const float *be[LAYERS], *bk[LAYERS], *bq[LAYERS], *bv[LAYERS];
  for (int l = 0; l < LAYERS; ++l) {
    int base = 14 + 8 * l;
    We[l] = (const float*)d_in[base + 0];
    Wk[l] = (const float*)d_in[base + 1];
    Wq[l] = (const float*)d_in[base + 2];
    Wv[l] = (const float*)d_in[base + 3];
    be[l] = (const float*)d_in[base + 4];
    bk[l] = (const float*)d_in[base + 5];
    bq[l] = (const float*)d_in[base + 6];
    bv[l] = (const float*)d_in[base + 7];
  }
  const float* emb = (const float*)d_in[46];
  const float* eW1 = (const float*)d_in[47];
  const float* eW2 = (const float*)d_in[48];
  const float* eb1 = (const float*)d_in[49];
  const float* eb2 = (const float*)d_in[50];
  const float* ebe1= (const float*)d_in[51];
  const float* eg1 = (const float*)d_in[52];
  const float* oW1 = (const float*)d_in[53];
  const float* oW2 = (const float*)d_in[54];
  const float* ob1 = (const float*)d_in[55];
  const float* ob2 = (const float*)d_in[56];
  const float* obe1= (const float*)d_in[57];
  const float* og1 = (const float*)d_in[58];
  float* out = (float*)d_out;

  // ---- workspace carve ----
  float* f = (float*)d_ws;
  float* h    = f; f += (size_t)N * 128;
  float* x0   = f; f += (size_t)N * 128;
  float* qb   = f; f += (size_t)N * 128;
  float* kb   = f; f += (size_t)N * 128;
  float* vb   = f; f += (size_t)N * 128;
  float* agg  = f; f += (size_t)N * 128;   // also reused as h1 in emb MLP
  float* alpha= f; f += (size_t)E * 4;     // reused in-place as exp()
  unsigned* mx = (unsigned*)f; f += (size_t)N * 4;
  float* den  = f; f += (size_t)N * 4;
  float* te   = f; f += 32 * 128;
  float* t1   = f; f += 32 * 128;
  float* ssum = f; f += 128;
  float* ssq  = f; f += 128;
  float* meanv= f; f += 128;
  float* rstdv= f; f += 128;
  float* pooled = f; f += (size_t)G * 128;
  float* cnt  = f; f += G;
  float* tmp2 = f; f += (size_t)G * 128;

  const int tilesN = (N + 63) / 64;                  // 782
  const int gridN  = tilesN < 256 ? tilesN : 256;    // persistent blocks
  const int tilesG = (G + 63) / 64;                  // 8
  const int edgeWaveBlocks = (E * 32 + 255) / 256;
  const int nodeElemBlocks = (N * 128 + 255) / 256;

  // ---- embedding + emb_mlp: h = relu(bn(T1[x])) @ W2 + b2 ; x0 = h ----
  table_gemm<<<1, 128, 0, stream>>>(emb, eW1, eb1, t1, 30);
  lookup_rows<<<nodeElemBlocks, 256, 0, stream>>>(xs, t1, agg, N);
  fill_k<<<1, 256, 0, stream>>>((unsigned*)ssum, 0u, 256);
  colstats_k<<<256, 128, 0, stream>>>(agg, N, ssum, ssq);
  finalize_k<<<1, 128, 0, stream>>>(ssum, ssq, 1.0f / N, meanv, rstdv);
  gemm128_wmma<<<gridN, 128, 0, stream>>>(agg, eW2, eb2, meanv, rstdv,
                                          eg1, ebe1, 2, h, x0, N, tilesN);

  // ---- conv layers ----
  for (int l = 0; l < LAYERS; ++l) {
    table_gemm<<<1, 128, 0, stream>>>(emb, We[l], be[l], te, 30);
    gemm128_wmma<<<gridN, 128, 0, stream>>>(h, Wq[l], bq[l],
        nullptr, nullptr, nullptr, nullptr, 0, qb, nullptr, N, tilesN);
    gemm128_wmma<<<gridN, 128, 0, stream>>>(h, Wk[l], bk[l],
        nullptr, nullptr, nullptr, nullptr, 0, kb, nullptr, N, tilesN);
    gemm128_wmma<<<gridN, 128, 0, stream>>>(h, Wv[l], bv[l],
        nullptr, nullptr, nullptr, nullptr, 0, vb, nullptr, N, tilesN);

    fill_k<<<(N*4 + 255)/256, 256, 0, stream>>>(mx, 0x007FFFFFu, N * 4);  // enc(-inf)
    fill_k<<<(N*4 + 255)/256, 256, 0, stream>>>((unsigned*)den, 0u, N * 4);
    fill_k<<<nodeElemBlocks, 256, 0, stream>>>((unsigned*)agg, 0u, N * 128);

    edge_alpha_k<<<edgeWaveBlocks, 256, 0, stream>>>(qb, kb, te, src, dst, eattr,
                                                     alpha, mx, E);
    edge_exp_k<<<(E*4 + 255)/256, 256, 0, stream>>>(dst, mx, alpha, den, E);
    edge_agg_k<<<edgeWaveBlocks, 256, 0, stream>>>(vb, te, src, dst, eattr,
                                                   alpha, den, agg, E);

    fill_k<<<1, 256, 0, stream>>>((unsigned*)ssum, 0u, 256);
    colstats_k<<<256, 128, 0, stream>>>(agg, N, ssum, ssq);
    finalize_k<<<1, 128, 0, stream>>>(ssum, ssq, 1.0f / N, meanv, rstdv);
    bn_relu_res_k<<<nodeElemBlocks, 256, 0, stream>>>(agg, meanv, rstdv,
        bns_g[l], bns_b[l], x0, h, N);
  }

  // ---- global mean pool + bn_end + out_mlp ----
  fill_k<<<(G*128 + 255)/256, 256, 0, stream>>>((unsigned*)pooled, 0u, G * 128);
  fill_k<<<(G + 255)/256, 256, 0, stream>>>((unsigned*)cnt, 0u, G);
  pool_sum_k<<<(N*32 + 255)/256, 256, 0, stream>>>(h, batch, pooled, cnt, N);
  pool_div_k<<<(G*128 + 255)/256, 256, 0, stream>>>(pooled, cnt, G);

  fill_k<<<1, 256, 0, stream>>>((unsigned*)ssum, 0u, 256);
  colstats_k<<<256, 128, 0, stream>>>(pooled, G, ssum, ssq);
  finalize_k<<<1, 128, 0, stream>>>(ssum, ssq, 1.0f / G, meanv, rstdv);
  // tmp2 = bn_end(pooled) @ oW1 + ob1   (BN on A-load, no ReLU)
  gemm128_wmma<<<tilesG, 128, 0, stream>>>(pooled, oW1, ob1, meanv, rstdv,
                                           bnend_g, bnend_b, 1, tmp2, nullptr, G, tilesG);
  fill_k<<<1, 256, 0, stream>>>((unsigned*)ssum, 0u, 256);
  colstats_k<<<256, 128, 0, stream>>>(tmp2, G, ssum, ssq);
  finalize_k<<<1, 128, 0, stream>>>(ssum, ssq, 1.0f / G, meanv, rstdv);
  out_final_k<<<G, 128, 0, stream>>>(tmp2, meanv, rstdv, og1, obe1, oW2, ob2, out);
}
